// TemplateSegmentAssembler_65859028517424
// MI455X (gfx1250) — matrix operand order repaired and checked by reading, not compiled, not run on hardware
//
#include <hip/hip_runtime.h>
#include <math.h>

// ---------------- problem constants (from the reference) ----------------
static constexpr int N_   = 160;
static constexpr int C_   = 512;
static constexpr int H_   = 256;   // hidden
static constexpr int L_   = 3;
static constexpr int H2_  = 512;   // 2H  (msg MLP mid width)
static constexpr int H4_  = 1024;  // 4H  (update MLP mid width)
static constexpr int KIN_ = 516;   // 2H + E
static constexpr int NC_  = N_ * C_;

typedef __bf16 bf16;
typedef __bf16 v16bf __attribute__((ext_vector_type(16)));
typedef __bf16 v8bf  __attribute__((ext_vector_type(8)));
typedef float  v8f   __attribute__((ext_vector_type(8)));

// ---------------- small helpers ----------------
__device__ __forceinline__ bf16 f2bf(float f) {
  union { float f; unsigned u; } in; in.f = f;
  unsigned u = in.u;
  u += 0x7FFFu + ((u >> 16) & 1u);          // round-to-nearest-even
  union { unsigned short s; bf16 b; } out;
  out.s = (unsigned short)(u >> 16);
  return out.b;
}
__device__ __forceinline__ float bf2f(bf16 b) {
  union { bf16 b; unsigned short s; } in; in.b = b;
  union { unsigned u; float f; } out; out.u = ((unsigned)in.s) << 16;
  return out.f;
}
__device__ __forceinline__ float gelu_exact(float x) {
  return 0.5f * x * (1.0f + erff(x * 0.70710678118654752f));
}

// A fragment: 16x32 bf16, row-major (lda multiple of 8, 16B-aligned base).
// Lane layout (ISA): m = lane&15 ; K = {half*8..half*8+7 , 16+half*8..16+half*8+7}
__device__ __forceinline__ v16bf load_a_frag(const bf16* A, int lda, int lane) {
  int half = lane >> 4, m = lane & 15;
  const bf16* p = A + (size_t)m * lda + half * 8;
  v8bf lo = *(const v8bf*)p;
  v8bf hi = *(const v8bf*)(p + 16);
  return __builtin_shufflevector(lo, hi, 0, 1, 2, 3, 4, 5, 6, 7,
                                 8, 9, 10, 11, 12, 13, 14, 15);
}
// B fragment from PRE-TRANSPOSED weights BT (N x K row-major, ldk = K).
// Lane layout (ISA): n = lane&15 ; K = {half*8..half*8+7 , 16+half*8..16+half*8+7}
__device__ __forceinline__ v16bf load_bt_frag(const bf16* BT, int ldk, int lane) {
  int half = lane >> 4, n = lane & 15;
  const bf16* p = BT + (size_t)n * ldk + half * 8;
  v8bf lo = *(const v8bf*)p;
  v8bf hi = *(const v8bf*)(p + 16);
  return __builtin_shufflevector(lo, hi, 0, 1, 2, 3, 4, 5, 6, 7,
                                 8, 9, 10, 11, 12, 13, 14, 15);
}
__device__ __forceinline__ v8f wmma_bf16(v16bf a, v16bf b, v8f c) {
  return __builtin_amdgcn_wmma_f32_16x16x32_bf16(false, a, false, b, (short)0, c, false, false);
}

// ---------------- kernels ----------------

__global__ void cvt_bf16_kernel(const float* __restrict__ src, bf16* __restrict__ dst, int n) {
  int i = blockIdx.x * blockDim.x + threadIdx.x;
  if (i < n) dst[i] = f2bf(src[i]);
}

// transpose-convert: src f32 (K x N row-major) -> dst bf16 (N x K row-major)
__global__ void tcvt_kernel(const float* __restrict__ src, bf16* __restrict__ dst, int K, int N) {
  int i = blockIdx.x * blockDim.x + threadIdx.x;
  if (i < K * N) {
    int k = i / N, n = i - k * N;
    dst[(size_t)n * K + k] = f2bf(src[i]);
  }
}

// LayerNorm over H=256, one row per 256-thread block, bf16 output
__global__ void ln_kernel(const float* __restrict__ x, const float* __restrict__ g,
                          const float* __restrict__ b, bf16* __restrict__ out) {
  __shared__ float rs[8], rs2[8];
  int row = blockIdx.x, t = threadIdx.x;
  float v = x[(size_t)row * H_ + t];
  float s = v, s2 = v * v;
#pragma unroll
  for (int o = 16; o > 0; o >>= 1) { s += __shfl_down(s, o, 32); s2 += __shfl_down(s2, o, 32); }
  int lane = t & 31, w = t >> 5;
  if (lane == 0) { rs[w] = s; rs2[w] = s2; }
  __syncthreads();
  if (t == 0) {
    float a = 0.f, a2 = 0.f;
#pragma unroll
    for (int i = 0; i < 8; ++i) { a += rs[i]; a2 += rs2[i]; }
    rs[0] = a; rs2[0] = a2;
  }
  __syncthreads();
  float mean = rs[0] * (1.0f / H_);
  float var  = rs2[0] * (1.0f / H_) - mean * mean;
  float inv  = rsqrtf(var + 1e-5f);
  out[(size_t)row * H_ + t] = f2bf((v - mean) * inv * g[t] + b[t]);
}

// G = A(bf16 MxK) @ BT'(bf16, stored N x K) -> bf16; BM=BN=64, 8 waves.
// Wave w owns row-block (w&3), col-blocks (w>>2)*32 and +16: one A-frag feeds 2 WMMAs.
__global__ void gemm_g_kernel(const bf16* __restrict__ A, const bf16* __restrict__ BT,
                              bf16* __restrict__ Cout, int K, int lda, int ldc) {
  int bm = blockIdx.x * 64, bn = blockIdx.y * 64;
  int w = threadIdx.x >> 5, lane = threadIdx.x & 31;
  int wm = (w & 3) * 16, wn0 = (w >> 2) * 32;
  v8f acc0 = {}, acc1 = {};
  for (int k0 = 0; k0 < K; k0 += 32) {
    v16bf af = load_a_frag(A + (size_t)(bm + wm) * lda + k0, lda, lane);
    v16bf b0 = load_bt_frag(BT + (size_t)(bn + wn0) * K + k0, K, lane);
    v16bf b1 = load_bt_frag(BT + (size_t)(bn + wn0 + 16) * K + k0, K, lane);
    acc0 = wmma_bf16(af, b0, acc0);
    acc1 = wmma_bf16(af, b1, acc1);
  }
  int nc  = lane & 15;
  int mbase = bm + wm + ((lane >> 4) << 3);
#pragma unroll
  for (int r = 0; r < 8; ++r) {
    Cout[(size_t)(mbase + r) * ldc + (bn + wn0 + nc)]      = f2bf(acc0[r]);
    Cout[(size_t)(mbase + r) * ldc + (bn + wn0 + 16 + nc)] = f2bf(acc1[r]);
  }
}

// Fused message kernel for one offset.  Each block owns output rows r = c0..c0+15 of
// sequence n, and computes BOTH the forward message of pair (r, r+off) and the reverse
// message of pair (r-off, r), so agg updates need no atomics (deterministic).
__global__ void msg_kernel(const bf16* __restrict__ G1, const bf16* __restrict__ G2,
                           const float* __restrict__ xyz, const int* __restrict__ valid,
                           const bf16* __restrict__ W1g,   // geo rows 512..515 (row-major, ld=512)
                           const float* __restrict__ b1,
                           const bf16* __restrict__ W2T,   // transposed: 256 x 512
                           const float* __restrict__ b2,
                           float* __restrict__ agg, int off) {
  __shared__ __align__(16) bf16 actF[16][H2_];   // 16 KB
  __shared__ __align__(16) bf16 actR[16][H2_];   // 16 KB
  __shared__ __align__(16) bf16 geoW[4][H2_];    // 4 KB
  __shared__ float b1s[H2_];                     // 2 KB
  __shared__ float sux[2][16], suy[2][16], suz[2][16], sd[2][16];
  __shared__ float svF[16], svR[16];

  const int n  = blockIdx.y;
  const int c0 = blockIdx.x * 16;
  const int t  = threadIdx.x;
  const int Cm = C_ - off;

  for (int e = t; e < 4 * H2_; e += 256) geoW[e >> 9][e & (H2_ - 1)] = W1g[e];
  for (int e = t; e < H2_; e += 256)     b1s[e] = b1[e];

  if (t < 32) {
    int dir = t >> 4, r = t & 15;
    int c = c0 + r;
    bool ok; int ci;
    if (dir == 0) { ok = (c < Cm);  ci = ok ? c : (Cm - 1); }     // fwd pair (ci, ci+off)
    else          { ok = (c >= off); ci = ok ? (c - off) : 0; }   // rev pair (ci, ci+off), j = row
    size_t i = (size_t)n * C_ + ci, j = i + off;
    float dx = xyz[j * 3 + 0] - xyz[i * 3 + 0];
    float dy = xyz[j * 3 + 1] - xyz[i * 3 + 1];
    float dz = xyz[j * 3 + 2] - xyz[i * 3 + 2];
    float d = fmaxf(sqrtf(dx * dx + dy * dy + dz * dz), 1e-6f);
    sux[dir][r] = dx / d; suy[dir][r] = dy / d; suz[dir][r] = dz / d; sd[dir][r] = d;
    float v = (ok && valid[i] && valid[j]) ? 1.f : 0.f;
    if (dir == 0) svF[r] = v; else svR[r] = v;
  }
  __syncthreads();

  // pre-activation (shared G1/G2 algebra) + exact gelu -> bf16 act in LDS
  for (int e = t; e < 16 * H2_; e += 256) {
    int r = e >> 9, q = e & (H2_ - 1);
    int c = c0 + r;
    float w0 = bf2f(geoW[0][q]), w1 = bf2f(geoW[1][q]);
    float w2 = bf2f(geoW[2][q]), w3 = bf2f(geoW[3][q]);
    float bb = b1s[q];
    {   // forward: pair (c, c+off) -> concat[h_in[i], h_in[j], unit, dist]
      int ci = (c < Cm) ? c : (Cm - 1);
      size_t i = (size_t)n * C_ + ci, j = i + off;
      float dird = sux[0][r] * w0 + suy[0][r] * w1 + suz[0][r] * w2;
      float p = bf2f(G1[i * H2_ + q]) + bf2f(G2[j * H2_ + q]) + dird + sd[0][r] * w3 + bb;
      actF[r][q] = f2bf(gelu_exact(p));
    }
    {   // reverse: pair (c-off, c) -> concat[h_in[j], h_in[i], -unit, dist]
      int ci = (c >= off) ? (c - off) : 0;
      size_t i = (size_t)n * C_ + ci, j = i + off;
      float dird = sux[1][r] * w0 + suy[1][r] * w1 + suz[1][r] * w2;
      float p = bf2f(G1[j * H2_ + q]) + bf2f(G2[i * H2_ + q]) - dird + sd[1][r] * w3 + bb;
      actR[r][q] = f2bf(gelu_exact(p));
    }
  }
  __syncthreads();

  // second GEMM (K=512, N=256): wave w owns n-tiles w*32 and w*32+16.
  // Per k-step: 2 LDS A-frags (fwd/rev) + 2 global B-frags feed 4 WMMAs.
  int w = t >> 5, lane = t & 31;
  int n0 = w * 32;
  v8f aF[2] = {}, aR[2] = {};
  for (int k0 = 0; k0 < H2_; k0 += 32) {
    v16bf fa = load_a_frag(&actF[0][0] + k0, H2_, lane);
    v16bf ra = load_a_frag(&actR[0][0] + k0, H2_, lane);
#pragma unroll
    for (int it = 0; it < 2; ++it) {
      v16bf bfrag = load_bt_frag(W2T + (size_t)(n0 + it * 16) * H2_ + k0, H2_, lane);
      aF[it] = wmma_bf16(fa, bfrag, aF[it]);
      aR[it] = wmma_bf16(ra, bfrag, aR[it]);
    }
  }
#pragma unroll
  for (int it = 0; it < 2; ++it) {
    int col = n0 + it * 16 + (lane & 15);
    int mb = (lane >> 4) << 3;
    float bias = b2[col];
#pragma unroll
    for (int r = 0; r < 8; ++r) {
      int m = mb + r;
      size_t row = (size_t)n * C_ + c0 + m;
      float val = svF[m] * (aF[it][r] + bias) + svR[m] * (aR[it][r] + bias);
      agg[row * H_ + col] += val;   // block owns these rows: no atomics needed
    }
  }
}

// h = h + agg*vf  (in place), then LN(h; ug, ub) -> bf16
__global__ void resid_ln_kernel(float* __restrict__ h, const float* __restrict__ agg,
                                const int* __restrict__ valid,
                                const float* __restrict__ g, const float* __restrict__ b,
                                bf16* __restrict__ out) {
  __shared__ float rs[8], rs2[8];
  int row = blockIdx.x, t = threadIdx.x;
  float vf = valid[row] ? 1.f : 0.f;
  float v = h[(size_t)row * H_ + t] + agg[(size_t)row * H_ + t] * vf;
  h[(size_t)row * H_ + t] = v;
  float s = v, s2 = v * v;
#pragma unroll
  for (int o = 16; o > 0; o >>= 1) { s += __shfl_down(s, o, 32); s2 += __shfl_down(s2, o, 32); }
  int lane = t & 31, w = t >> 5;
  if (lane == 0) { rs[w] = s; rs2[w] = s2; }
  __syncthreads();
  if (t == 0) {
    float a = 0.f, a2 = 0.f;
#pragma unroll
    for (int i = 0; i < 8; ++i) { a += rs[i]; a2 += rs2[i]; }
    rs[0] = a; rs2[0] = a2;
  }
  __syncthreads();
  float mean = rs[0] * (1.0f / H_);
  float var  = rs2[0] * (1.0f / H_) - mean * mean;
  float inv  = rsqrtf(var + 1e-5f);
  out[(size_t)row * H_ + t] = f2bf((v - mean) * inv * g[t] + b[t]);
}

// Fused update MLP: out = (h + gelu(t @ uW1 + ub1) @ uW2 + ub2) * vf
__global__ void update_kernel(const bf16* __restrict__ tin, const float* __restrict__ h,
                              const int* __restrict__ valid,
                              const bf16* __restrict__ uW1T,  // transposed: 1024 x 256
                              const float* __restrict__ ub1,
                              const bf16* __restrict__ uW2T,  // transposed: 256 x 1024
                              const float* __restrict__ ub2,
                              float* __restrict__ out) {
  __shared__ __align__(16) bf16 a1[16][H_];    // 8 KB
  __shared__ __align__(16) bf16 act[16][H4_];  // 32 KB
  int m0 = blockIdx.x * 16;
  int t = threadIdx.x, w = t >> 5, lane = t & 31;

  for (int e = t; e < 16 * H_; e += 256) {
    int r = e >> 8, q = e & (H_ - 1);
    a1[r][q] = tin[(size_t)(m0 + r) * H_ + q];
  }
  __syncthreads();

  // GEMM1: 256 -> 1024, 64 n-tiles, 8 per wave; one A-frag per k-step feeds 8 WMMAs
  {
    v8f acc[8] = {};
    for (int k0 = 0; k0 < H_; k0 += 32) {
      v16bf af = load_a_frag(&a1[0][0] + k0, H_, lane);
#pragma unroll
      for (int it = 0; it < 8; ++it) {
        int n0 = (w * 8 + it) * 16;
        v16bf bfrag = load_bt_frag(uW1T + (size_t)n0 * H_ + k0, H_, lane);
        acc[it] = wmma_bf16(af, bfrag, acc[it]);
      }
    }
#pragma unroll
    for (int it = 0; it < 8; ++it) {
      int col = (w * 8 + it) * 16 + (lane & 15);
      int mb = (lane >> 4) << 3;
      float bias = ub1[col];
#pragma unroll
      for (int r = 0; r < 8; ++r)
        act[mb + r][col] = f2bf(gelu_exact(acc[it][r] + bias));
    }
  }
  __syncthreads();

  // GEMM2: 1024 -> 256, 16 n-tiles, 2 per wave; one A-frag per k-step feeds 2 WMMAs
  {
    v8f acc[2] = {};
    for (int k0 = 0; k0 < H4_; k0 += 32) {
      v16bf af = load_a_frag(&act[0][0] + k0, H4_, lane);
#pragma unroll
      for (int it = 0; it < 2; ++it) {
        int n0 = (w * 2 + it) * 16;
        v16bf bfrag = load_bt_frag(uW2T + (size_t)n0 * H4_ + k0, H4_, lane);
        acc[it] = wmma_bf16(af, bfrag, acc[it]);
      }
    }
#pragma unroll
    for (int it = 0; it < 2; ++it) {
      int col = (w * 2 + it) * 16 + (lane & 15);
      int mb = (lane >> 4) << 3;
      float bias = ub2[col];
#pragma unroll
      for (int r = 0; r < 8; ++r) {
        int row = m0 + mb + r;
        float vf = valid[row] ? 1.f : 0.f;
        float u = h[(size_t)row * H_ + col] + acc[it][r] + bias;
        out[(size_t)row * H_ + col] = u * vf;
      }
    }
  }
}

// ---------------- host ----------------
extern "C" void kernel_launch(void* const* d_in, const int* in_sizes, int n_in,
                              void* d_out, int out_size, void* d_ws, size_t ws_size,
                              hipStream_t stream) {
  const float* h0    = (const float*)d_in[0];
  const float* xyz   = (const float*)d_in[1];
  const int*   valid = (const int*)d_in[2];
  const float* nn_g  = (const float*)d_in[3];
  const float* nn_b  = (const float*)d_in[4];
  const float* W1    = (const float*)d_in[5];
  const float* b1    = (const float*)d_in[6];
  const float* W2    = (const float*)d_in[7];
  const float* b2    = (const float*)d_in[8];
  const float* ug    = (const float*)d_in[9];
  const float* ub    = (const float*)d_in[10];
  const float* uW1   = (const float*)d_in[11];
  const float* ub1   = (const float*)d_in[12];
  const float* uW2   = (const float*)d_in[13];
  const float* ub2   = (const float*)d_in[14];
  (void)in_sizes; (void)n_in; (void)out_size; (void)ws_size;

  char* ws = (char*)d_ws;
  size_t off_b = 0;
  auto alloc = [&](size_t bytes) -> char* {
    char* p = ws + off_b;
    off_b = (off_b + bytes + 255) & ~(size_t)255;
    return p;
  };
  float* hbuf  = (float*)alloc((size_t)NC_ * H_ * 4);
  float* agg   = (float*)alloc((size_t)NC_ * H_ * 4);
  bf16* hin    = (bf16*)alloc((size_t)NC_ * H_ * 2);
  bf16* tbf    = (bf16*)alloc((size_t)NC_ * H_ * 2);
  bf16* G1     = (bf16*)alloc((size_t)NC_ * H2_ * 2);
  bf16* G2     = (bf16*)alloc((size_t)NC_ * H2_ * 2);
  bf16* W1aT   = (bf16*)alloc((size_t)L_ * H2_ * H_ * 2);   // (512 x 256) per layer
  bf16* W1bT   = (bf16*)alloc((size_t)L_ * H2_ * H_ * 2);   // (512 x 256) per layer
  bf16* W1gbf  = (bf16*)alloc((size_t)L_ * 4 * H2_ * 2);    // geo rows, row-major
  bf16* W2T    = (bf16*)alloc((size_t)L_ * H_ * H2_ * 2);   // (256 x 512) per layer
  bf16* uW1T   = (bf16*)alloc((size_t)L_ * H4_ * H_ * 2);   // (1024 x 256) per layer
  bf16* uW2T   = (bf16*)alloc((size_t)L_ * H_ * H4_ * 2);   // (256 x 1024) per layer

  for (int l = 0; l < L_; ++l) {
    const float* W1l = W1 + (size_t)l * KIN_ * H2_;
    int nKN;
    nKN = H_ * H2_;
    tcvt_kernel<<<(nKN + 255) / 256, 256, 0, stream>>>(W1l, W1aT + (size_t)l * H2_ * H_, H_, H2_);
    tcvt_kernel<<<(nKN + 255) / 256, 256, 0, stream>>>(W1l + (size_t)H_ * H2_,
                                                       W1bT + (size_t)l * H2_ * H_, H_, H2_);
    cvt_bf16_kernel<<<(4 * H2_ + 255) / 256, 256, 0, stream>>>(W1l + (size_t)2 * H_ * H2_,
                                                               W1gbf + (size_t)l * 4 * H2_, 4 * H2_);
    nKN = H2_ * H_;
    tcvt_kernel<<<(nKN + 255) / 256, 256, 0, stream>>>(W2 + (size_t)l * H2_ * H_,
                                                       W2T + (size_t)l * H_ * H2_, H2_, H_);
    nKN = H_ * H4_;
    tcvt_kernel<<<(nKN + 255) / 256, 256, 0, stream>>>(uW1 + (size_t)l * H_ * H4_,
                                                       uW1T + (size_t)l * H4_ * H_, H_, H4_);
    nKN = H4_ * H_;
    tcvt_kernel<<<(nKN + 255) / 256, 256, 0, stream>>>(uW2 + (size_t)l * H4_ * H_,
                                                       uW2T + (size_t)l * H_ * H4_, H4_, H_);
  }
  hipMemcpyAsync(hbuf, h0, (size_t)NC_ * H_ * 4, hipMemcpyDeviceToDevice, stream);

  const int offs[5] = {1, 2, 4, 8, 16};
  for (int l = 0; l < L_; ++l) {
    ln_kernel<<<NC_, 256, 0, stream>>>(hbuf, nn_g + l * H_, nn_b + l * H_, hin);

    dim3 gg(NC_ / 64, H2_ / 64);
    gemm_g_kernel<<<gg, 256, 0, stream>>>(hin, W1aT + (size_t)l * H2_ * H_, G1, H_, H_, H2_);
    gemm_g_kernel<<<gg, 256, 0, stream>>>(hin, W1bT + (size_t)l * H2_ * H_, G2, H_, H_, H2_);

    hipMemsetAsync(agg, 0, (size_t)NC_ * H_ * 4, stream);
    for (int oi = 0; oi < 5; ++oi) {
      dim3 mg(C_ / 16, N_);
      msg_kernel<<<mg, 256, 0, stream>>>(G1, G2, xyz, valid,
                                         W1gbf + (size_t)l * 4 * H2_,
                                         b1 + l * H2_,
                                         W2T + (size_t)l * H_ * H2_, b2 + l * H_,
                                         agg, offs[oi]);
    }

    resid_ln_kernel<<<NC_, 256, 0, stream>>>(hbuf, agg, valid, ug + l * H_, ub + l * H_, tbf);

    float* outp = (l == L_ - 1) ? (float*)d_out : hbuf;
    update_kernel<<<NC_ / 16, 256, 0, stream>>>(tbf, hbuf, valid,
                                                uW1T + (size_t)l * H4_ * H_, ub1 + l * H4_,
                                                uW2T + (size_t)l * H_ * H4_, ub2 + l * H_,
                                                outp);
  }
}
